// Attention_1924145348930
// MI455X (gfx1250) — matrix operand order repaired and checked by reading, not compiled, not run on hardware
//
#include <hip/hip_runtime.h>
#include <hip/hip_bf16.h>

// ---------------------------------------------------------------------------
// MHA for MI455X (gfx1250, wave32). All matmuls via v_wmma_f32_16x16x32_bf16.
// Roofline: ~69 GFLOP vs ~200 MB traffic (mask alone is 134 MB) -> matrix
// pipes for all GEMMs; mask streamed non-temporally and staged via LDS so the
// dominant byte stream costs 1 b128 VMEM op per 16 bytes instead of 16 scalar
// u8 loads.
// ---------------------------------------------------------------------------

#define D_MODEL 1024
#define NHEAD   16
#define DH      64
#define BATCH   2
#define SEQ     2048
#define ROWS    (BATCH*SEQ)      // 4096

typedef __attribute__((ext_vector_type(16))) __bf16 v16bf;
typedef __attribute__((ext_vector_type(8)))  __bf16 v8bf;
typedef __attribute__((ext_vector_type(8)))  float  v8f;
typedef __attribute__((ext_vector_type(4)))  unsigned int v4u;

union ABFrag { v16bf v; v8bf h[2]; };

static __device__ __forceinline__ unsigned short f2bf(float f) {
  unsigned int u = __float_as_uint(f);
  u += 0x7FFFu + ((u >> 16) & 1u);          // round-to-nearest-even
  return (unsigned short)(u >> 16);
}

static __device__ __forceinline__ v8f wmma_bf16(v16bf a, v16bf b, v8f c) {
  // (neg_a, A, neg_b, B, c_mod, C, reuse_a, reuse_b)
  return __builtin_amdgcn_wmma_f32_16x16x32_bf16(false, a, false, b,
                                                 (short)0, c, false, false);
}

// ---------------------------------------------------------------------------
// 1) elementwise f32 -> bf16
// ---------------------------------------------------------------------------
__global__ void cvt_f32_bf16(const float* __restrict__ in,
                             unsigned short* __restrict__ out, int n4) {
  int i = blockIdx.x * blockDim.x + threadIdx.x;
  if (i >= n4) return;
  float4 f = ((const float4*)in)[i];
  ushort4 o;
  o.x = f2bf(f.x); o.y = f2bf(f.y); o.z = f2bf(f.z); o.w = f2bf(f.w);
  ((ushort4*)out)[i] = o;
}

// ---------------------------------------------------------------------------
// 2) W[k][n] f32 -> Wt[n][k] bf16 (LDS 32x32 tile transpose)
// ---------------------------------------------------------------------------
__global__ void cvt_transpose_w(const float* __restrict__ W,
                                unsigned short* __restrict__ Wt) {
  __shared__ float tile[32][33];
  const int tx = threadIdx.x, ty = threadIdx.y;
  const int n0 = blockIdx.x * 32, k0 = blockIdx.y * 32;
#pragma unroll
  for (int i = 0; i < 4; i++)
    tile[ty + 8 * i][tx] = W[(size_t)(k0 + ty + 8 * i) * D_MODEL + n0 + tx];
  __syncthreads();
#pragma unroll
  for (int i = 0; i < 4; i++)
    Wt[(size_t)(n0 + ty + 8 * i) * D_MODEL + k0 + tx] =
        f2bf(tile[tx][ty + 8 * i]);
}

// ---------------------------------------------------------------------------
// 3) WMMA GEMM: C[4096,1024] = A(bf16, row-major) x B (given as Bt[n][k] bf16)
//    Block tile 128x64, 8 waves in 4x2 grid, wave tile 32x32, K-step 32.
//    mode 0: f32 row-major to outF (final Wo projection)
//    mode 1: bf16 head-major [hb][L][dh]   (Q, K)
//    mode 2: bf16 head-major transposed [hb][dh][L]  (V)
// ---------------------------------------------------------------------------
#define BM 128
#define BN 64
#define BK 32
#define SA 40   // padded bf16 stride (80B rows, 16B-aligned chunks)
#define SB 40

__global__ __launch_bounds__(256)
void gemm_bf16(const unsigned short* __restrict__ A,
               const unsigned short* __restrict__ Bt,
               float* __restrict__ outF,
               unsigned short* __restrict__ outH,
               int mode) {
  __shared__ unsigned short As[BM * SA];
  __shared__ unsigned short Bs[BN * SB];
  const int t = threadIdx.x;
  const int lane = t & 31, w = t >> 5;
  const int wy = w & 3, wx = w >> 2;
  const int lm = lane & 15, lh = lane >> 4;
  const int m0 = blockIdx.y * BM;
  const int n0 = blockIdx.x * BN;

  v8f acc[2][2] = {};

  for (int k0 = 0; k0 < D_MODEL; k0 += BK) {
    __syncthreads();
    // A tile: 128 rows x 32 k = 512 16-byte chunks, 2 per thread
#pragma unroll
    for (int i = 0; i < 2; i++) {
      int c = t + 256 * i;
      int r = c >> 2, kc = (c & 3) * 8;
      *(v8bf*)&As[r * SA + kc] =
          *(const v8bf*)&A[(size_t)(m0 + r) * D_MODEL + k0 + kc];
    }
    // Bt tile: 64 n x 32 k = 256 chunks, 1 per thread
    {
      int r = t >> 2, kc = (t & 3) * 8;
      *(v8bf*)&Bs[r * SB + kc] =
          *(const v8bf*)&Bt[(size_t)(n0 + r) * D_MODEL + k0 + kc];
    }
    if (k0 + BK < D_MODEL) {   // gfx1250 global_prefetch_b8 of next tiles
      __builtin_prefetch(&A[(size_t)(m0 + (t >> 1)) * D_MODEL + k0 + BK], 0, 1);
      __builtin_prefetch(&Bt[(size_t)(n0 + (t & 63)) * D_MODEL + k0 + BK], 0, 1);
    }
    __syncthreads();

    ABFrag a[2], b[2];
#pragma unroll
    for (int i = 0; i < 2; i++) {   // A frag: lane holds row lm, K=(lh*8)+{0..7}, 16+lh*8+{0..7}
      int r = wy * 32 + i * 16 + lm;
      a[i].h[0] = *(const v8bf*)&As[r * SA + lh * 8];
      a[i].h[1] = *(const v8bf*)&As[r * SA + 16 + lh * 8];
    }
#pragma unroll
    for (int j = 0; j < 2; j++) {   // B frag: lane holds col lm, K=(lh*8)+{0..7}, 16+lh*8+{0..7}
      int n = wx * 32 + j * 16 + lm;
      b[j].h[0] = *(const v8bf*)&Bs[n * SB + lh * 8];
      b[j].h[1] = *(const v8bf*)&Bs[n * SB + 16 + lh * 8];
    }
#pragma unroll
    for (int i = 0; i < 2; i++)
#pragma unroll
      for (int j = 0; j < 2; j++)
        acc[i][j] = wmma_bf16(a[i].v, b[j].v, acc[i][j]);
  }

  // epilogue: C element (r, lane) -> M = r + 8*lh, N = lm  (ISA C layout)
#pragma unroll
  for (int i = 0; i < 2; i++) {
#pragma unroll
    for (int j = 0; j < 2; j++) {
#pragma unroll
      for (int r = 0; r < 8; r++) {
        int M = m0 + wy * 32 + i * 16 + r + 8 * lh;
        int N = n0 + wx * 32 + j * 16 + lm;
        float v = acc[i][j][r];
        if (mode == 0) {
          // output written once, never re-read by us -> non-temporal
          __builtin_nontemporal_store(v, &outF[(size_t)M * D_MODEL + N]);
        } else {
          int h = N >> 6, d = N & 63;
          int b_ = M >> 11, q = M & (SEQ - 1);
          int hb = h * BATCH + b_;
          size_t idx = (mode == 1)
                           ? ((size_t)hb * SEQ + q) * DH + d
                           : ((size_t)hb * DH + d) * SEQ + q;
          outH[idx] = f2bf(v);
        }
      }
    }
  }
}

// ---------------------------------------------------------------------------
// 4) Flash attention. Grid: (SEQ/128, H*B), 256 threads = 8 waves.
//    Each wave owns a 16-row q-tile; 64-key chunks of K, V^T and the 128x64
//    mask tile staged in LDS each iteration.
// ---------------------------------------------------------------------------
#define SKV 72   // padded bf16 stride for 64-wide LDS tiles (144B rows)

__global__ __launch_bounds__(256)
void attn_kernel(const unsigned short* __restrict__ Q,   // [32][2048][64]
                 const unsigned short* __restrict__ K,   // [32][2048][64]
                 const unsigned short* __restrict__ VT,  // [32][64][2048]
                 const unsigned char* __restrict__ mask, // [32][2048][2048]
                 unsigned short* __restrict__ ctx) {     // [4096][1024]
  __shared__ unsigned short Ks[64 * SKV];
  __shared__ unsigned short Vs[64 * SKV];
  __shared__ unsigned short Ps[8 * 16 * SKV];            // per-wave P staging
  __shared__ __attribute__((aligned(16))) unsigned char Ms[128 * 64]; // mask tile

  const int t = threadIdx.x, lane = t & 31, w = t >> 5;
  const int lm = lane & 15, lh = lane >> 4;
  const int hb = blockIdx.y;
  const int qblk = blockIdx.x * 128;
  const int q0 = qblk + w * 16;
  const float scale = 0.125f;               // 1/sqrt(dh=64)
  const float NEGF = -2147483647.0f;        // 1 - 2^31

  // Q fragments for this wave's 16 rows, kept in registers for whole kernel
  ABFrag qf[2];
  {
    const unsigned short* Qr = Q + ((size_t)hb * SEQ + q0 + lm) * DH;
#pragma unroll
    for (int s = 0; s < 2; s++) {
      qf[s].h[0] = *(const v8bf*)&Qr[s * 32 + lh * 8];
      qf[s].h[1] = *(const v8bf*)&Qr[s * 32 + 16 + lh * 8];
    }
  }

  float row_m[8], row_l[8];
#pragma unroll
  for (int r = 0; r < 8; r++) { row_m[r] = -__builtin_inff(); row_l[r] = 0.f; }
  v8f acc[4] = {};

  unsigned short* Pw = &Ps[w * 16 * SKV];
  const unsigned char* mblk = mask + ((size_t)hb * SEQ + qblk) * SEQ;

  for (int kk0 = 0; kk0 < SEQ; kk0 += 64) {
    __syncthreads();
    // stage K chunk [64 key][64 d] and V^T chunk [64 d][64 key]
#pragma unroll
    for (int i = 0; i < 2; i++) {
      int c = t + 256 * i;
      int r = c >> 3, kc = (c & 7) * 8;
      *(v8bf*)&Ks[r * SKV + kc] =
          *(const v8bf*)&K[((size_t)hb * SEQ + kk0 + r) * DH + kc];
      *(v8bf*)&Vs[r * SKV + kc] =
          *(const v8bf*)&VT[((size_t)hb * DH + r) * SEQ + kk0 + kc];
    }
    // stage mask tile [128 q][64 key]: 512 16B chunks, 2 per thread,
    // non-temporal (134 MB stream read exactly once; keep it out of L2 LRU)
#pragma unroll
    for (int i = 0; i < 2; i++) {
      int c = t + 256 * i;
      int r = c >> 2, kc = (c & 3) * 16;
      v4u m = __builtin_nontemporal_load(
          (const v4u*)&mblk[(size_t)r * SEQ + kk0 + kc]);
      *(v4u*)&Ms[r * 64 + kc] = m;
    }
    __syncthreads();

    // scores: 4 subtiles of 16 keys, K-dim = dh in two 32-steps
    v8f sc[4];
#pragma unroll
    for (int sub = 0; sub < 4; sub++) {
      v8f s = {};
#pragma unroll
      for (int st = 0; st < 2; st++) {
        ABFrag kb;
        int n = sub * 16 + lm;
        kb.h[0] = *(const v8bf*)&Ks[n * SKV + st * 32 + lh * 8];
        kb.h[1] = *(const v8bf*)&Ks[n * SKV + st * 32 + 16 + lh * 8];
        s = wmma_bf16(qf[st].v, kb.v, s);
      }
      sc[sub] = s;
    }

    // mask + scale (bytes now come from LDS, not global)
#pragma unroll
    for (int sub = 0; sub < 4; sub++) {
#pragma unroll
      for (int r = 0; r < 8; r++) {
        int qloc = r + 8 * lh;
        unsigned char mv = Ms[(w * 16 + qloc) * 64 + sub * 16 + lm];
        float v = sc[sub][r];
        sc[sub][r] = mv ? v * scale : NEGF;
      }
    }

    // online softmax; row stats live per-(r, half-wave), replicated in 16 lanes
    float alpha[8];
#pragma unroll
    for (int r = 0; r < 8; r++) {
      float m = fmaxf(fmaxf(sc[0][r], sc[1][r]), fmaxf(sc[2][r], sc[3][r]));
#pragma unroll
      for (int off = 8; off >= 1; off >>= 1)
        m = fmaxf(m, __shfl_xor(m, off, 32));
      float mn = fmaxf(row_m[r], m);
      alpha[r] = __expf(row_m[r] - mn);
      row_m[r] = mn;
      float ls = 0.f;
#pragma unroll
      for (int sub = 0; sub < 4; sub++) {
        float p = __expf(sc[sub][r] - mn);
        sc[sub][r] = p;
        ls += p;
      }
#pragma unroll
      for (int off = 8; off >= 1; off >>= 1)
        ls += __shfl_xor(ls, off, 32);
      row_l[r] = row_l[r] * alpha[r] + ls;
    }

    // rescale running context accumulator
#pragma unroll
    for (int n = 0; n < 4; n++)
#pragma unroll
      for (int r = 0; r < 8; r++)
        acc[n][r] *= alpha[r];

    // P (C-frag layout) -> bf16 -> per-wave LDS (A-frag layout re-read)
#pragma unroll
    for (int sub = 0; sub < 4; sub++)
#pragma unroll
      for (int r = 0; r < 8; r++)
        Pw[(r + 8 * lh) * SKV + sub * 16 + lm] = f2bf(sc[sub][r]);
    asm volatile("s_wait_dscnt 0" ::: "memory");  // LDS in-order per wave

    // ctx += P x V  (V^T rows are kk-contiguous -> contiguous B-fragments)
#pragma unroll
    for (int st = 0; st < 2; st++) {
      ABFrag pf;
      pf.h[0] = *(const v8bf*)&Pw[lm * SKV + st * 32 + lh * 8];
      pf.h[1] = *(const v8bf*)&Pw[lm * SKV + st * 32 + 16 + lh * 8];
#pragma unroll
      for (int n = 0; n < 4; n++) {
        ABFrag vb;
        int d = n * 16 + lm;
        vb.h[0] = *(const v8bf*)&Vs[d * SKV + st * 32 + lh * 8];
        vb.h[1] = *(const v8bf*)&Vs[d * SKV + st * 32 + 16 + lh * 8];
        acc[n] = wmma_bf16(pf.v, vb.v, acc[n]);
      }
    }
  }

  // normalize and write ctx (bf16) head-concatenated: [b*L + q][h*64 + d]
  const int h = hb >> 1, b_ = hb & 1;
#pragma unroll
  for (int n = 0; n < 4; n++) {
#pragma unroll
    for (int r = 0; r < 8; r++) {
      int q = q0 + r + 8 * lh;
      int d = n * 16 + lm;
      float v = acc[n][r] / row_l[r];
      ctx[((size_t)(b_ * SEQ + q)) * D_MODEL + h * DH + d] = f2bf(v);
    }
  }
}

// ---------------------------------------------------------------------------
// launch
// ---------------------------------------------------------------------------
extern "C" void kernel_launch(void* const* d_in, const int* in_sizes, int n_in,
                              void* d_out, int out_size, void* d_ws,
                              size_t ws_size, hipStream_t stream) {
  const float* x = (const float*)d_in[0];
  const float* y = (const float*)d_in[1];
  const unsigned char* mask = (const unsigned char*)d_in[2];  // jnp.bool_ = 1B
  const float* Wq = (const float*)d_in[3];
  const float* Wk = (const float*)d_in[4];
  const float* Wv = (const float*)d_in[5];
  const float* Wo = (const float*)d_in[6];
  float* out = (float*)d_out;

  char* ws = (char*)d_ws;
  size_t off = 0;
  auto alloc = [&](size_t bytes) -> void* {
    void* p = ws + off;
    off += (bytes + 255) & ~(size_t)255;
    return p;
  };
  const size_t XN = (size_t)ROWS * D_MODEL;          // 4096*1024
  unsigned short* Xb  = (unsigned short*)alloc(XN * 2);
  unsigned short* Yb  = (unsigned short*)alloc(XN * 2);
  unsigned short* WqT = (unsigned short*)alloc((size_t)D_MODEL * D_MODEL * 2);
  unsigned short* WkT = (unsigned short*)alloc((size_t)D_MODEL * D_MODEL * 2);
  unsigned short* WvT = (unsigned short*)alloc((size_t)D_MODEL * D_MODEL * 2);
  unsigned short* WoT = (unsigned short*)alloc((size_t)D_MODEL * D_MODEL * 2);
  unsigned short* Qb  = (unsigned short*)alloc(XN * 2);  // [32][2048][64]
  unsigned short* Kb  = (unsigned short*)alloc(XN * 2);  // [32][2048][64]
  unsigned short* VTb = (unsigned short*)alloc(XN * 2);  // [32][64][2048]
  unsigned short* CTX = (unsigned short*)alloc(XN * 2);  // [4096][1024]
  (void)ws_size; (void)in_sizes; (void)n_in; (void)out_size;

  // 1) activations -> bf16
  int n4 = (int)(XN / 4);
  cvt_f32_bf16<<<dim3((n4 + 255) / 256), dim3(256), 0, stream>>>(x, Xb, n4);
  cvt_f32_bf16<<<dim3((n4 + 255) / 256), dim3(256), 0, stream>>>(y, Yb, n4);

  // 2) weights -> transposed bf16 (Bt[n][k])
  dim3 tg(D_MODEL / 32, D_MODEL / 32), tb(32, 8);
  cvt_transpose_w<<<tg, tb, 0, stream>>>(Wq, WqT);
  cvt_transpose_w<<<tg, tb, 0, stream>>>(Wk, WkT);
  cvt_transpose_w<<<tg, tb, 0, stream>>>(Wv, WvT);
  cvt_transpose_w<<<tg, tb, 0, stream>>>(Wo, WoT);

  // 3) projections
  dim3 gg(D_MODEL / BN, ROWS / BM), gb(256);
  gemm_bf16<<<gg, gb, 0, stream>>>(Xb, WqT, nullptr, Qb, 1);   // Q head-major
  gemm_bf16<<<gg, gb, 0, stream>>>(Yb, WkT, nullptr, Kb, 1);   // K head-major
  gemm_bf16<<<gg, gb, 0, stream>>>(Yb, WvT, nullptr, VTb, 2);  // V^T

  // 4) attention
  attn_kernel<<<dim3(SEQ / 128, NHEAD * BATCH), dim3(256), 0, stream>>>(
      Qb, Kb, VTb, mask, CTX);

  // 5) output projection -> f32
  gemm_bf16<<<gg, gb, 0, stream>>>(CTX, WoT, out, nullptr, 0);
}
